// RLxLSTM_22591527977493
// MI455X (gfx1250) — matrix operand order repaired and checked by reading, not compiled, not run on hardware
//
#include <hip/hip_runtime.h>
#include <hip/hip_bf16.h>
#include <stdint.h>

// ---------------- problem constants ----------------
constexpr int   B_  = 64;
constexpr int   S_  = 512;
constexpr int   I_  = 512;
constexpr int   H_  = 1024;
constexpr int   NH_ = 4;
constexpr int   D_  = 256;
constexpr float EPS_ = 1e-5f;

// ---------------- vector types (CDNA5 WMMA operands) ----------------
typedef __bf16 bf16_t;
typedef __attribute__((ext_vector_type(16))) __bf16 v16bf;
typedef __attribute__((ext_vector_type(8)))  __bf16 v8bf;
typedef __attribute__((ext_vector_type(4)))  __bf16 v4bf;
typedef __attribute__((ext_vector_type(8)))  float  v8f;

// ---------------- WMMA fragment loaders ----------------
// bf16 A-matrix 16x32 lane layout (ISA 7.12.2):
//   lane l<16 : row=l,  K in {k0..k0+7} U {k0+16..k0+23}
//   lane l>=16: row=l-16, K in {k0+8..k0+15} U {k0+24..k0+31}
__device__ __forceinline__ v16bf make_v16(v8bf lo, v8bf hi) {
    return __builtin_shufflevector(lo, hi, 0,1,2,3,4,5,6,7,8,9,10,11,12,13,14,15);
}

__device__ __forceinline__ v16bf load_a_frag(const bf16_t* __restrict__ A, int lda,
                                             int lane, int k0) {
    const int r    = lane & 15;
    const int ksub = (lane & 16) ? 8 : 0;
    const bf16_t* p = A + (size_t)r * lda + k0 + ksub;
    return make_v16(*(const v8bf*)p, *(const v8bf*)(p + 16));
}

// bf16 B-matrix 32x16: lane<16 holds K in {k0..k0+7} U {k0+16..k0+23} of column (lane&15),
// lane>=16 holds K in {k0+8..k0+15} U {k0+24..k0+31}.  Bm is stored (N x K) row-major => contiguous.
__device__ __forceinline__ v16bf load_b_frag(const bf16_t* __restrict__ Bm, int ldb,
                                             int lane, int k0) {
    const int c    = lane & 15;
    const int ksub = (lane & 16) ? 8 : 0;
    const bf16_t* p = Bm + (size_t)c * ldb + k0 + ksub;
    return make_v16(*(const v8bf*)p, *(const v8bf*)(p + 16));
}

__device__ __forceinline__ v8f wmma_bf16(v16bf a, v16bf b, v8f c) {
    return __builtin_amdgcn_wmma_f32_16x16x32_bf16(false, a, false, b,
                                                   (short)0, c, false, false);
}

__device__ __forceinline__ float sigmoid_f(float x) { return 1.f / (1.f + __expf(-x)); }
__device__ __forceinline__ float logsig_f(float x) {
    return fminf(x, 0.f) - log1pf(__expf(-fabsf(x)));
}

// ---------------- async copy: global -> LDS (ASYNCcnt-tracked) ----------------
// Low 32 bits of a generic shared pointer == wave-relative LDS byte address.
__device__ __forceinline__ void async_g2l_b128(uint32_t lds_off, const void* gptr) {
    asm volatile("global_load_async_to_lds_b128 %0, %1, off"
                 :: "v"(lds_off), "v"(gptr) : "memory");
}

// ---------------- K0a: fp32 -> bf16 elementwise ----------------
__global__ void __launch_bounds__(256) convert_kernel(const float* __restrict__ src,
                                                      bf16_t* __restrict__ dst, size_t n) {
    size_t i = (size_t)blockIdx.x * blockDim.x + threadIdx.x;
    size_t stride = (size_t)gridDim.x * blockDim.x;
    for (; i < n; i += stride) dst[i] = (bf16_t)src[i];
}

// ---------------- K0b: [M][D][D] (d,e) -> bf16 [M][D][D] (e,d) ----------------
__global__ void __launch_bounds__(256) convert_T_kernel(const float* __restrict__ src,
                                                        bf16_t* __restrict__ dst, int nmat) {
    size_t total = (size_t)nmat * D_ * D_;
    size_t i = (size_t)blockIdx.x * blockDim.x + threadIdx.x;
    if (i >= total) return;
    int e   = (int)(i % D_);
    int d   = (int)((i / D_) % D_);
    int mtx = (int)(i / ((size_t)D_ * D_));
    dst[((size_t)mtx * D_ + e) * D_ + d] = (bf16_t)src[i];
}

// ---------------- K0c: flag = all(state0.n == 0) ----------------
__global__ void __launch_bounds__(1024) nflag_kernel(const float* __restrict__ state0,
                                                     int* __restrict__ flag) {
    __shared__ int ok;
    const int tid = threadIdx.x;
    if (tid == 0) ok = 1;
    __syncthreads();
    bool all0 = true;
    const float* nptr = state0 + (size_t)2 * B_ * H_;
    for (int i = tid; i < B_ * H_; i += 1024) all0 &= (nptr[i] == 0.f);
    if (!all0) atomicAnd(&ok, 0);
    __syncthreads();
    if (tid == 0) *flag = ok;
}

// ---------------- K1: xp = x @ Wproj^T + bproj  (bf16 WMMA, fp32 acc) ----------
__global__ void __launch_bounds__(256) proj_gemm_kernel(const bf16_t* __restrict__ xbf,
                                                        const bf16_t* __restrict__ wbf,
                                                        const float* __restrict__ bproj,
                                                        float* __restrict__ xp) {
    const int lane = threadIdx.x & 31;
    const int wid  = threadIdx.x >> 5;
    const int m0 = blockIdx.x * 128 + (wid & 1) * 64;
    const int n0 = blockIdx.y * 128 + (wid >> 1) * 32;

    v8f acc[4][2];
#pragma unroll
    for (int i = 0; i < 4; ++i)
#pragma unroll
        for (int j = 0; j < 2; ++j)
#pragma unroll
            for (int r = 0; r < 8; ++r) acc[i][j][r] = 0.f;

    for (int k0 = 0; k0 < I_; k0 += 32) {
        v16bf a[4];
#pragma unroll
        for (int i = 0; i < 4; ++i)
            a[i] = load_a_frag(xbf + (size_t)(m0 + i * 16) * I_, I_, lane, k0);
#pragma unroll
        for (int j = 0; j < 2; ++j) {
            v16bf bfr = load_b_frag(wbf + (size_t)(n0 + j * 16) * I_, I_, lane, k0);
#pragma unroll
            for (int i = 0; i < 4; ++i) acc[i][j] = wmma_bf16(a[i], bfr, acc[i][j]);
        }
    }
    const int rb = (lane & 16) ? 8 : 0;
    const int cl = lane & 15;
#pragma unroll
    for (int i = 0; i < 4; ++i)
#pragma unroll
        for (int j = 0; j < 2; ++j) {
            const int col = n0 + j * 16 + cl;
            const float bias = bproj[col];
#pragma unroll
            for (int r = 0; r < 8; ++r) {
                const int row = m0 + i * 16 + rb + r;
                xp[(size_t)row * H_ + col] = acc[i][j][r] + bias;
            }
        }
}

// ---------------- K2: row LayerNorm (scale only), fp32 -> bf16 ----------------
__global__ void __launch_bounds__(256) ln_rows_kernel(const float* __restrict__ xp,
                                                      const float* __restrict__ w,
                                                      bf16_t* __restrict__ xn) {
    const int row = blockIdx.x;
    const int tid = threadIdx.x;
    __shared__ float s1[256], s2[256];
    const float4 v4 = *(const float4*)(xp + (size_t)row * H_ + tid * 4);
    float va[4] = {v4.x, v4.y, v4.z, v4.w};
    float s = 0.f, q = 0.f;
#pragma unroll
    for (int k = 0; k < 4; ++k) { s += va[k]; q += va[k] * va[k]; }
    s1[tid] = s; s2[tid] = q;
    __syncthreads();
    for (int off = 128; off > 0; off >>= 1) {
        if (tid < off) { s1[tid] += s1[tid + off]; s2[tid] += s2[tid + off]; }
        __syncthreads();
    }
    const float mu  = s1[0] * (1.f / H_);
    const float var = s2[0] * (1.f / H_) - mu * mu;
    const float rs  = rsqrtf(var + EPS_);
    v4bf o;
#pragma unroll
    for (int k = 0; k < 4; ++k) {
        const int col = tid * 4 + k;
        o[k] = (bf16_t)((va[k] - mu) * rs * w[col]);
    }
    *(v4bf*)(xn + (size_t)row * H_ + tid * 4) = o;
}

// ---------------- K3: gates_in = xn_head @ Wg^T(e,d)  -> [S][NH][4][B][D] -----
__global__ void __launch_bounds__(256) gates_gemm_kernel(const bf16_t* __restrict__ xn,
                                                         const bf16_t* __restrict__ wgT,
                                                         float* __restrict__ gates) {
    const int lane = threadIdx.x & 31;
    const int wid  = threadIdx.x >> 5;
    const int gh = blockIdx.z;
    const int g  = gh >> 2;
    const int h  = gh & 3;
    const int m0 = blockIdx.x * 128 + (wid & 1) * 64;
    const int n0 = blockIdx.y * 128 + (wid >> 1) * 32;
    const bf16_t* A  = xn + (size_t)h * D_;                       // lda = H_
    const bf16_t* Bm = wgT + (size_t)(g * NH_ + h) * D_ * D_;     // ldb = D_

    v8f acc[4][2];
#pragma unroll
    for (int i = 0; i < 4; ++i)
#pragma unroll
        for (int j = 0; j < 2; ++j)
#pragma unroll
            for (int r = 0; r < 8; ++r) acc[i][j][r] = 0.f;

    for (int k0 = 0; k0 < D_; k0 += 32) {
        v16bf a[4];
#pragma unroll
        for (int i = 0; i < 4; ++i)
            a[i] = load_a_frag(A + (size_t)(m0 + i * 16) * H_, H_, lane, k0);
#pragma unroll
        for (int j = 0; j < 2; ++j) {
            v16bf bfr = load_b_frag(Bm + (size_t)(n0 + j * 16) * D_, D_, lane, k0);
#pragma unroll
            for (int i = 0; i < 4; ++i) acc[i][j] = wmma_bf16(a[i], bfr, acc[i][j]);
        }
    }
    const int rb = (lane & 16) ? 8 : 0;
    const int cl = lane & 15;
#pragma unroll
    for (int i = 0; i < 4; ++i)
#pragma unroll
        for (int r = 0; r < 8; ++r) {
            const int m = m0 + i * 16 + rb + r;
            const int b = m >> 9;          // m = b*S + s
            const int s = m & (S_ - 1);
            float* orow = gates + ((((size_t)s * NH_ + h) * 4 + g) * B_ + b) * D_;
#pragma unroll
            for (int j = 0; j < 2; ++j) orow[n0 + j * 16 + cl] = acc[i][j][r];
        }
}

// ---------------- K4: sequential sLSTM scan ----------------
// Batch rows are independent through the recurrence (R couples only d->e within
// a row), so grid = (NH heads) x (B/16 batch groups) = 16 persistent WGs.
// 1024 threads = 32 waves; wave (g = wid>>3) computes a 16x32 slice of ry.
// LDS: y 16x256 bf16 (8KB) + ry 4x16x256 f32 (64KB)
//      + double-buffered gate block 2 x 4x16x256 f32 (128KB) = 200KB.
// Gates for step t+1 are fetched with global_load_async_to_lds_b128 (ASYNCcnt)
// overlapped with the step-t WMMA work; s_wait_asynccnt 0x4 retires the older
// slot (4 in-order async ops per wave per slot) before the barrier.
__global__ void __launch_bounds__(1024) scan_kernel(const float* __restrict__ gates,
                                                    const bf16_t* __restrict__ rT,
                                                    const float* __restrict__ bcell,
                                                    const float* __restrict__ state0,
                                                    const int* __restrict__ flag,
                                                    float* __restrict__ ys,
                                                    float* __restrict__ fstate) {
    const int h    = blockIdx.x;
    const int b0   = blockIdx.y * 16;
    const int tid  = threadIdx.x;
    const int lane = tid & 31;
    const int wid  = tid >> 5;

    extern __shared__ char smem[];
    bf16_t* yls   = (bf16_t*)smem;                         // [16][256] bf16
    float*  rawls = (float*)(smem + 16 * 256 * 2);         // [4][16][256] f32
    char*   gbuf  = smem + 16 * 256 * 2 + 4 * 16 * 256 * 4;// 2 x 64KB gate slots
    const uint32_t gbuf_lds = (uint32_t)(uintptr_t)gbuf;   // wave-relative LDS addr

    // elementwise ownership: local row rloc = tid>>6, 4 contiguous cols
    const int rloc = tid >> 6;
    const int e0   = (tid & 63) * 4;
    const int b    = b0 + rloc;
    float yv[4], cv[4], nv[4], mv[4], bc[4][4];
#pragma unroll
    for (int k = 0; k < 4; ++k) {
        const int col = h * D_ + e0 + k;
        yv[k] = state0[(size_t)0 * B_ * H_ + (size_t)b * H_ + col];
        cv[k] = state0[(size_t)1 * B_ * H_ + (size_t)b * H_ + col];
        nv[k] = state0[(size_t)2 * B_ * H_ + (size_t)b * H_ + col];
        mv[k] = state0[(size_t)3 * B_ * H_ + (size_t)b * H_ + col];
        yls[rloc * 256 + e0 + k] = (bf16_t)yv[k];
#pragma unroll
        for (int g = 0; g < 4; ++g)
            bc[g][k] = bcell[(size_t)(g * NH_ + h) * D_ + e0 + k];
    }
    const bool flag0 = (*flag != 0);

    // prologue: async-stage gates for t=0 into slot 0 (4 ops per wave)
#pragma unroll
    for (int q = 0; q < 4; ++q) {
        const int u   = (q << 10) + tid;           // 0..4095 float4 units
        const int gg  = u >> 10;                   // gate
        const int rem = u & 1023;                  // unit within [16][256]
        const float* gp = gates + (((size_t)h * 4 + gg) * B_ + b0) * D_ + (size_t)rem * 4;
        async_g2l_b128(gbuf_lds + ((uint32_t)u << 4), gp);
    }
    __syncthreads();

    // GEMM mapping: gate g = wid>>3, 32-col slice nsub = (wid&7)*32
    const int g    = wid >> 3;
    const int nsub = (wid & 7) * 32;
    const bf16_t* Rg = rT + (size_t)(g * NH_ + h) * D_ * D_;   // (e,d) row-major

    for (int t = 0; t < S_; ++t) {
        // async-stage gates for t+1 into the other slot (overlaps the GEMM)
        {
            const int tn = (t + 1 < S_) ? (t + 1) : t;   // final-step issue is a dummy
            const int slot = (t + 1) & 1;
            const size_t tb = ((size_t)tn * NH_ + h) * 4;
#pragma unroll
            for (int q = 0; q < 4; ++q) {
                const int u   = (q << 10) + tid;
                const int gg  = u >> 10;
                const int rem = u & 1023;
                const float* gp = gates + ((tb + gg) * B_ + b0) * D_ + (size_t)rem * 4;
                async_g2l_b128(gbuf_lds + ((uint32_t)slot << 16) + ((uint32_t)u << 4), gp);
            }
        }

        // ry = y_{t-1} @ R_g : 16 x 32 slice, K = 256
        v8f acc[2];
#pragma unroll
        for (int j = 0; j < 2; ++j)
#pragma unroll
            for (int r = 0; r < 8; ++r) acc[j][r] = 0.f;

#pragma unroll
        for (int ks = 0; ks < 8; ++ks) {
            const int k0 = ks * 32;
            const v16bf a = load_a_frag(yls, 256, lane, k0);
#pragma unroll
            for (int j = 0; j < 2; ++j) {
                v16bf bfr = load_b_frag(Rg + (size_t)(nsub + j * 16) * D_, D_, lane, k0);
                acc[j] = wmma_bf16(a, bfr, acc[j]);
            }
        }
        const int rb = (lane & 16) ? 8 : 0;
        const int cl = lane & 15;
#pragma unroll
        for (int j = 0; j < 2; ++j)
#pragma unroll
            for (int r = 0; r < 8; ++r)
                rawls[((size_t)g * 16 + rb + r) * 256 + nsub + j * 16 + cl] = acc[j][r];

        // retire slot-t async ops (per-wave, in order: keep the 4 newest in flight)
        asm volatile("s_wait_asynccnt 0x4" ::: "memory");
        __syncthreads();

        const float* gb = (const float*)(gbuf + ((size_t)(t & 1) << 16));
        const bool first = (t == 0) && flag0;
#pragma unroll
        for (int k = 0; k < 4; ++k) {
            const int e = e0 + k;
            const float ir  = rawls[((size_t)0 * 16 + rloc) * 256 + e] +
                              gb[((size_t)0 * 16 + rloc) * 256 + e] + bc[0][k];
            const float fr  = rawls[((size_t)1 * 16 + rloc) * 256 + e] +
                              gb[((size_t)1 * 16 + rloc) * 256 + e] + bc[1][k];
            const float zr  = rawls[((size_t)2 * 16 + rloc) * 256 + e] +
                              gb[((size_t)2 * 16 + rloc) * 256 + e] + bc[2][k];
            const float orw = rawls[((size_t)3 * 16 + rloc) * 256 + e] +
                              gb[((size_t)3 * 16 + rloc) * 256 + e] + bc[3][k];
            const float lfm  = mv[k] + logsig_f(fr);
            const float mnew = first ? ir : fmaxf(ir, lfm);
            const float og = sigmoid_f(orw);
            const float ig = __expf(ir - mnew);
            const float fg = __expf(lfm - mnew);
            const float cn = fg * cv[k] + ig * tanhf(zr);
            const float nn = fg * nv[k] + ig;
            const float yn = og * cn / nn;
            cv[k] = cn; nv[k] = nn; mv[k] = mnew; yv[k] = yn;
        }
        // publish y_t (bf16 in LDS for next step's A-operand) and stream to HBM
        v4bf yo;
#pragma unroll
        for (int k = 0; k < 4; ++k) yo[k] = (bf16_t)yv[k];
        *(v4bf*)(yls + rloc * 256 + e0) = yo;
        float4 yf = make_float4(yv[0], yv[1], yv[2], yv[3]);
        *(float4*)(ys + (((size_t)b * S_ + t) * NH_ + h) * D_ + e0) = yf;
        __syncthreads();
    }

    // final_state = (y, c, n, m) stacked, shape (4, B, H)
#pragma unroll
    for (int k = 0; k < 4; ++k) {
        const int col = h * D_ + e0 + k;
        fstate[(size_t)0 * B_ * H_ + (size_t)b * H_ + col] = yv[k];
        fstate[(size_t)1 * B_ * H_ + (size_t)b * H_ + col] = cv[k];
        fstate[(size_t)2 * B_ * H_ + (size_t)b * H_ + col] = nv[k];
        fstate[(size_t)3 * B_ * H_ + (size_t)b * H_ + col] = mv[k];
    }
}

// ---------------- K5: head groupnorm + residual + post-LN ---------------------
__global__ void __launch_bounds__(256) finalize_kernel(const float* __restrict__ xp,
                                                       const float* __restrict__ ys,
                                                       const float* __restrict__ gn_w,
                                                       const float* __restrict__ lnp_w,
                                                       float* __restrict__ out) {
    const int row = blockIdx.x;   // b*S + s
    const int tid = threadIdx.x;
    __shared__ float s1[256], s2[256];

    const float4 y4 = *(const float4*)(ys + (size_t)row * H_ + tid * 4);
    float ya[4] = {y4.x, y4.y, y4.z, y4.w};
    float s = 0.f, q = 0.f;
#pragma unroll
    for (int k = 0; k < 4; ++k) { s += ya[k]; q += ya[k] * ya[k]; }
    s1[tid] = s; s2[tid] = q;
    __syncthreads();
    for (int off = 32; off > 0; off >>= 1) {
        if ((tid & 63) < off) { s1[tid] += s1[tid + off]; s2[tid] += s2[tid + off]; }
        __syncthreads();
    }
    const int base = tid & ~63;
    const float mu  = s1[base] * (1.f / D_);
    const float var = s2[base] * (1.f / D_) - mu * mu;
    const float rs  = rsqrtf(var + EPS_);
    __syncthreads();

    float tv[4]; float ls = 0.f, lq = 0.f;
#pragma unroll
    for (int k = 0; k < 4; ++k) {
        const int col = tid * 4 + k;
        const float yn = (ya[k] - mu) * rs * gn_w[col];
        const float v  = xp[(size_t)row * H_ + col] + yn;
        tv[k] = v; ls += v; lq += v * v;
    }
    s1[tid] = ls; s2[tid] = lq;
    __syncthreads();
    for (int off = 128; off > 0; off >>= 1) {
        if (tid < off) { s1[tid] += s1[tid + off]; s2[tid] += s2[tid + off]; }
        __syncthreads();
    }
    const float mu2  = s1[0] * (1.f / H_);
    const float var2 = s2[0] * (1.f / H_) - mu2 * mu2;
    const float rs2  = rsqrtf(var2 + EPS_);
#pragma unroll
    for (int k = 0; k < 4; ++k) {
        const int col = tid * 4 + k;
        out[(size_t)row * H_ + col] = (tv[k] - mu2) * rs2 * lnp_w[col];
    }
}

// ---------------- host-side launcher ----------------
extern "C" void kernel_launch(void* const* d_in, const int* in_sizes, int n_in,
                              void* d_out, int out_size, void* d_ws, size_t ws_size,
                              hipStream_t stream) {
    const float* x      = (const float*)d_in[0];
    const float* state0 = (const float*)d_in[1];
    const float* Wproj  = (const float*)d_in[2];
    const float* bproj  = (const float*)d_in[3];
    const float* ln1_w  = (const float*)d_in[4];
    const float* Wg     = (const float*)d_in[5];
    const float* R      = (const float*)d_in[6];
    const float* bcell  = (const float*)d_in[7];
    const float* gn_w   = (const float*)d_in[8];
    const float* lnp_w  = (const float*)d_in[9];

    float* out    = (float*)d_out;
    float* fstate = out + (size_t)B_ * S_ * H_;

    // workspace layout
    char* w = (char*)d_ws;
    bf16_t* xbf  = (bf16_t*)w;  w += (size_t)B_ * S_ * I_ * 2;
    bf16_t* wpbf = (bf16_t*)w;  w += (size_t)H_ * I_ * 2;
    bf16_t* wgT  = (bf16_t*)w;  w += (size_t)4 * NH_ * D_ * D_ * 2;
    bf16_t* rT   = (bf16_t*)w;  w += (size_t)4 * NH_ * D_ * D_ * 2;
    float*  xp   = (float*)w;   w += (size_t)B_ * S_ * H_ * 4;
    bf16_t* xn   = (bf16_t*)w;  w += (size_t)B_ * S_ * H_ * 2;
    float*  gate = (float*)w;   w += (size_t)S_ * NH_ * 4 * B_ * D_ * 4;
    float*  ysb  = (float*)w;   w += (size_t)B_ * S_ * H_ * 4;
    int*    flag = (int*)w;

    // K0: precision conversion (+ transposes for WMMA B-operand contiguity)
    {
        const size_t nx = (size_t)B_ * S_ * I_;
        convert_kernel<<<4096, 256, 0, stream>>>(x, xbf, nx);
        convert_kernel<<<2048, 256, 0, stream>>>(Wproj, wpbf, (size_t)H_ * I_);
        const int nmat = 4 * NH_;
        const int ntb = (nmat * D_ * D_ + 255) / 256;
        convert_T_kernel<<<ntb, 256, 0, stream>>>(Wg, wgT, nmat);
        convert_T_kernel<<<ntb, 256, 0, stream>>>(R, rT, nmat);
        nflag_kernel<<<1, 1024, 0, stream>>>(state0, flag);
    }

    // K1: input projection GEMM  (32768 x 1024 x 512)
    proj_gemm_kernel<<<dim3((B_ * S_) / 128, H_ / 128), 256, 0, stream>>>(xbf, wpbf, bproj, xp);

    // K2: pre-norm
    ln_rows_kernel<<<B_ * S_, 256, 0, stream>>>(xp, ln1_w, xn);

    // K3: gate projections (16 head-gate GEMMs, 32768 x 256 x 256 each)
    gates_gemm_kernel<<<dim3((B_ * S_) / 128, D_ / 128, 16), 256, 0, stream>>>(xn, wgT, gate);

    // K4: sequential scan, 16 persistent WGs (head x batch-group), 200KB LDS
    const size_t scan_lds = (size_t)16 * 256 * 2      // y (bf16)
                          + (size_t)4 * 16 * 256 * 4  // ry (f32)
                          + (size_t)2 * 4 * 16 * 256 * 4; // gate double buffer
    scan_kernel<<<dim3(NH_, B_ / 16), 1024, scan_lds, stream>>>(
        gate, rT, bcell, state0, flag, ysb, fstate);

    // K5: group-norm + residual + post-LN
    finalize_kernel<<<B_ * S_, 256, 0, stream>>>(xp, ysb, gn_w, lnp_w, out);
}
